// attention_propagantion_78262894068287
// MI455X (gfx1250) — matrix operand order repaired and checked by reading, not compiled, not run on hardware
//
#include <hip/hip_runtime.h>
#include <hip/hip_bf16.h>
#include <math.h>

// Problem constants (from the reference)
#define BB 2
#define CC 256
#define NN 8192
#define HH 4
#define KK 16
#define HD 64

typedef __attribute__((ext_vector_type(16))) _Float16 v16h;
typedef __attribute__((ext_vector_type(8)))  _Float16 v8h;
typedef __attribute__((ext_vector_type(8)))  float    v8f;

#define BP 40   // LDS pitch in halves: 80-B rows keep every 8-half chunk 16-B aligned

// ---------------------------------------------------------------------------
// WMMA GEMM, all-f16 inputs, K-contiguous ("n-major") activations:
//   Out[b][m][n] = epi( sum_ch W[m][ch] * X[b][n][ch] )
//   X channels < split come from X0 [b][n][split], rest from X1 (fused concat).
//   out_f32=1 -> store f32 [b][M][n] (+optional residual);  else f16 [b][n][M].
// Block: 128M x 64N, 8 waves as 4(M) x 2(N); wave = 32M x 32N = 4 WMMA/K-step.
// Both A (weights) and B (activations) tiles are staged to LDS with
// global_load_async_to_lds_b128 (ASYNCcnt), DOUBLE-BUFFERED: stage s+1 is in
// flight while stage s is consumed (s_wait_asynccnt 3 retires s, in-order).
// ---------------------------------------------------------------------------
__global__ __launch_bounds__(256)
void gemm_wmma_f16(const _Float16* __restrict__ Wh,
                   const _Float16* __restrict__ X0,
                   const _Float16* __restrict__ X1,
                   int split,
                   void* __restrict__ OutP, int out_f32,
                   const float* __restrict__ eshift,   // bias or BN shift (may be null)
                   const float* __restrict__ escale,   // BN scale (null -> 1.0)
                   const float* __restrict__ resid,    // residual f32 [b][M][N] (may be null)
                   int Mout, int Cin, int Npts, int relu)
{
    __shared__ __align__(16) _Float16 Alds[2][128 * BP];  // [row][kk]
    __shared__ __align__(16) _Float16 Blds[2][64 * BP];   // [col][kk]

    const int b    = blockIdx.z;
    const int m0   = blockIdx.y * 128;
    const int n0   = blockIdx.x * 64;
    const int tid  = threadIdx.x;
    const int lane = tid & 31;
    const int wave = tid >> 5;
    const int wmq  = wave >> 1;           // 0..3  -> 32-row group
    const int wnq  = wave & 1;            // 0..1  -> 32-col group

    // Staging chunks (8 halves = 16 B each):
    //  B: 64 cols x 32 K  = 256 chunks -> 1/thread   (col = tid>>2, k8 = (tid&3)*8)
    //  A: 128 rows x 32 K = 512 chunks -> 2/thread   (rows srow and srow+64)
    const int scol = tid >> 2;
    const int sk8  = (tid & 3) * 8;
    const int C1   = Cin - split;
    uint32_t ldsB[2], ldsA0[2], ldsA1[2];
#pragma unroll
    for (int s = 0; s < 2; ++s) {
        ldsB[s]  = (uint32_t)(size_t)(&Blds[s][scol * BP + sk8]);
        ldsA0[s] = (uint32_t)(size_t)(&Alds[s][scol * BP + sk8]);         // rows 0..63
        ldsA1[s] = (uint32_t)(size_t)(&Alds[s][(scol + 64) * BP + sk8]);  // rows 64..127
    }
    const _Float16* aSrc0 = Wh + (size_t)(m0 + scol) * Cin + sk8;
    const _Float16* aSrc1 = Wh + (size_t)(m0 + scol + 64) * Cin + sk8;

    v8f acc00 = {}, acc01 = {}, acc10 = {}, acc11 = {};
    const int nsteps = Cin >> 5;

    // ---- prologue: issue stage 0 into buffer 0 ----
    {
        const int ch = sk8;
        const _Float16* bsrc =
            (ch < split) ? (X0 + ((size_t)b * Npts + n0 + scol) * split + ch)
                         : (X1 + ((size_t)b * Npts + n0 + scol) * C1 + (ch - split));
        asm volatile("global_load_async_to_lds_b128 %0, %1, off"
                     :: "v"(ldsB[0]), "v"(bsrc) : "memory");
        asm volatile("global_load_async_to_lds_b128 %0, %1, off"
                     :: "v"(ldsA0[0]), "v"(aSrc0) : "memory");
        asm volatile("global_load_async_to_lds_b128 %0, %1, off"
                     :: "v"(ldsA1[0]), "v"(aSrc1) : "memory");
    }

    for (int s = 0; s < nsteps; ++s) {
        const int cur = s & 1;
        // ---- issue stage s+1 into the other buffer, then retire stage s ----
        if (s + 1 < nsteps) {
            const int nk0 = (s + 1) << 5;
            const int nxt = cur ^ 1;
            const int ch  = nk0 + sk8;
            const _Float16* bsrc =
                (ch < split) ? (X0 + ((size_t)b * Npts + n0 + scol) * split + ch)
                             : (X1 + ((size_t)b * Npts + n0 + scol) * C1 + (ch - split));
            asm volatile("global_load_async_to_lds_b128 %0, %1, off"
                         :: "v"(ldsB[nxt]), "v"(bsrc) : "memory");
            asm volatile("global_load_async_to_lds_b128 %0, %1, off"
                         :: "v"(ldsA0[nxt]), "v"(aSrc0 + nk0) : "memory");
            asm volatile("global_load_async_to_lds_b128 %0, %1, off"
                         :: "v"(ldsA1[nxt]), "v"(aSrc1 + nk0) : "memory");
            asm volatile("s_wait_asynccnt 0x3" ::: "memory");  // stage s retired
        } else {
            asm volatile("s_wait_asynccnt 0x0" ::: "memory");
        }
        __syncthreads();

        // ---- fragments from LDS (wave32 WMMA layouts) ----
        // A: lane<16 row=lane&15, halves0-7=K0..7, 8-15=K16..23; lane>=16 K+8
        const int kb  = (lane >> 4) * 8;
        const int kbb = (lane >> 4) * 8;
        v16h af0, af1, bf0, bf1;
        {
            const _Float16* ap = &Alds[cur][(wmq * 32 + (lane & 15)) * BP + kb];
            v8h x = *(const v8h*)(ap), y = *(const v8h*)(ap + 16);
#pragma unroll
            for (int i = 0; i < 8; ++i) { af0[i] = x[i]; af0[i + 8] = y[i]; }
        }
        {
            const _Float16* ap = &Alds[cur][(wmq * 32 + 16 + (lane & 15)) * BP + kb];
            v8h x = *(const v8h*)(ap), y = *(const v8h*)(ap + 16);
#pragma unroll
            for (int i = 0; i < 8; ++i) { af1[i] = x[i]; af1[i + 8] = y[i]; }
        }
        {
            const _Float16* bp = &Blds[cur][(wnq * 32 + (lane & 15)) * BP + kbb];
            v8h x = *(const v8h*)(bp), y = *(const v8h*)(bp + 16);
#pragma unroll
            for (int i = 0; i < 8; ++i) { bf0[i] = x[i]; bf0[i + 8] = y[i]; }
        }
        {
            const _Float16* bp = &Blds[cur][(wnq * 32 + 16 + (lane & 15)) * BP + kbb];
            v8h x = *(const v8h*)(bp), y = *(const v8h*)(bp + 16);
#pragma unroll
            for (int i = 0; i < 8; ++i) { bf1[i] = x[i]; bf1[i + 8] = y[i]; }
        }
        acc00 = __builtin_amdgcn_wmma_f32_16x16x32_f16(false, af0, false, bf0,
                                                       (short)0, acc00, false, false);
        acc01 = __builtin_amdgcn_wmma_f32_16x16x32_f16(false, af0, false, bf1,
                                                       (short)0, acc01, false, false);
        acc10 = __builtin_amdgcn_wmma_f32_16x16x32_f16(false, af1, false, bf0,
                                                       (short)0, acc10, false, false);
        acc11 = __builtin_amdgcn_wmma_f32_16x16x32_f16(false, af1, false, bf1,
                                                       (short)0, acc11, false, false);
        __syncthreads();   // readers done before buffer is overwritten again
    }

    // ---- epilogue ----
#pragma unroll
    for (int mt = 0; mt < 2; ++mt) {
        const int rbase = m0 + wmq * 32 + mt * 16 + 8 * (lane >> 4);
#pragma unroll
        for (int nt = 0; nt < 2; ++nt) {
            const int col = n0 + wnq * 32 + nt * 16 + (lane & 15);
            const v8f& acc = mt ? (nt ? acc11 : acc10) : (nt ? acc01 : acc00);
#pragma unroll
            for (int r = 0; r < 8; ++r) {
                const int row = rbase + r;
                float v  = acc[r];
                float sc = escale ? escale[row] : 1.0f;
                float sh = eshift ? eshift[row] : 0.0f;
                v = v * sc + sh;
                if (relu) v = fmaxf(v, 0.0f);
                if (out_f32) {
                    if (resid) v += resid[((size_t)b * Mout + row) * Npts + col];
                    ((float*)OutP)[((size_t)b * Mout + row) * Npts + col] = v;
                } else {
                    ((_Float16*)OutP)[((size_t)b * Npts + col) * Mout + row] = (_Float16)v;
                }
            }
        }
    }
}

// ---------------------------------------------------------------------------
// Neighbor attention, f16 in/out, [b][n][c] layout (c = h*64+d).
// One wave per (b,h,n); lanes hold d and d+32. oT may alias qT (each wave
// reads exactly the q slice it later writes).
// ---------------------------------------------------------------------------
__global__ __launch_bounds__(256)
void attn_kernel(const _Float16* __restrict__ qT, const _Float16* __restrict__ kT,
                 const _Float16* __restrict__ vT, _Float16* __restrict__ oT,
                 const int* __restrict__ neighb, const float* __restrict__ nmask,
                 const float* __restrict__ res_lam)
{
    const int wid  = blockIdx.x * 8 + (threadIdx.x >> 5);
    const int lane = threadIdx.x & 31;
    const int n    = wid % NN;
    const int bh   = wid / NN;
    const int h    = bh & (HH - 1);
    const int b    = bh >> 2;

    const _Float16* qp = qT + ((size_t)b * NN + n) * CC + h * HD;
    const float q0 = (float)qp[lane];
    const float q1 = (float)qp[lane + 32];
    const float lamh  = res_lam[h];
    const float scale = 0.125f;   // 1/sqrt(64)

    int   idx[KK];
    float ex[KK];
    float denom = 0.0f;

#pragma unroll
    for (int j = 0; j < KK; ++j) {
        const int id = neighb[((size_t)b * KK + j) * NN + n];
        idx[j] = id;
        const _Float16* kp = kT + ((size_t)b * NN + id) * CC + h * HD;
        float p = q0 * (float)kp[lane] + q1 * (float)kp[lane + 32];
#pragma unroll
        for (int s = 16; s > 0; s >>= 1) p += __shfl_xor(p, s, 32);
        float simi = p * scale + lamh * nmask[((size_t)b * KK + j) * NN + n];
        simi = fminf(fmaxf(simi, -30.0f), 30.0f);
        float e = __expf(simi);
        ex[j] = e;
        denom += e;
    }
    const float inv = 1.0f / (denom + 1e-8f);

    float o0 = 0.0f, o1 = 0.0f;
#pragma unroll
    for (int j = 0; j < KK; ++j) {
        const _Float16* vp = vT + ((size_t)b * NN + idx[j]) * CC + h * HD;
        const float s = ex[j] * inv;
        o0 += s * (float)vp[lane];
        o1 += s * (float)vp[lane + 32];
    }
    _Float16* op = oT + ((size_t)b * NN + n) * CC + h * HD;
    op[lane]      = (_Float16)o0;
    op[lane + 32] = (_Float16)o1;
}

// f32 [b][C][N] -> f16 [b][n][C] transpose-convert (LDS-tiled, both sides coalesced)
__global__ __launch_bounds__(256)
void cvt_transpose_kernel(const float* __restrict__ in, _Float16* __restrict__ out,
                          int Cc, int Nn)
{
    __shared__ float tile[32][33];
    const int b  = blockIdx.z;
    const int c0 = blockIdx.y * 32;
    const int n0 = blockIdx.x * 32;
    const int tx = threadIdx.x & 31;
    const int ty = threadIdx.x >> 5;      // 0..7
#pragma unroll
    for (int i = ty; i < 32; i += 8)
        tile[i][tx] = in[((size_t)b * Cc + c0 + i) * Nn + n0 + tx];
    __syncthreads();
#pragma unroll
    for (int i = ty; i < 32; i += 8)
        out[((size_t)b * Nn + n0 + i) * Cc + c0 + tx] = (_Float16)tile[tx][i];
}

// simple elementwise f32 -> f16 (weights)
__global__ void cvt_kernel(const float* __restrict__ in, _Float16* __restrict__ out, int n)
{
    int i = blockIdx.x * blockDim.x + threadIdx.x;
    if (i < n) out[i] = (_Float16)in[i];
}

// Fold BN into affine scale/shift: val = y*scale[m] + shift[m]
__global__ void bn_prep_kernel(const float* __restrict__ gamma,
                               const float* __restrict__ beta,
                               const float* __restrict__ mean,
                               const float* __restrict__ var,
                               float* __restrict__ scale,
                               float* __restrict__ shift, int n)
{
    int i = blockIdx.x * blockDim.x + threadIdx.x;
    if (i < n) {
        float s  = gamma[i] * rsqrtf(var[i] + 1e-5f);
        scale[i] = s;
        shift[i] = beta[i] - mean[i] * s;
    }
}

extern "C" void kernel_launch(void* const* d_in, const int* in_sizes, int n_in,
                              void* d_out, int out_size, void* d_ws, size_t ws_size,
                              hipStream_t stream) {
    const float* desc1  = (const float*)d_in[0];
    const float* desc2  = (const float*)d_in[1];
    const int*   neighb = (const int*)  d_in[2];
    const float* nmask  = (const float*)d_in[3];
    // d_in[4] res_attention: unused by the reference
    const float* res_lam = (const float*)d_in[5];
    const float* wq = (const float*)d_in[6];  const float* bq = (const float*)d_in[7];
    const float* wk = (const float*)d_in[8];  const float* bk = (const float*)d_in[9];
    const float* wv = (const float*)d_in[10]; const float* bv = (const float*)d_in[11];
    const float* wm = (const float*)d_in[12]; const float* bm = (const float*)d_in[13];
    const float* wc1      = (const float*)d_in[14];
    const float* bn_gamma = (const float*)d_in[15];
    const float* bn_beta  = (const float*)d_in[16];
    const float* bn_mean  = (const float*)d_in[17];
    const float* bn_var   = (const float*)d_in[18];
    const float* wc2 = (const float*)d_in[19]; const float* bc2 = (const float*)d_in[20];

    _Float16* H = (_Float16*)d_ws;
    const size_t T = (size_t)BB * NN * CC;     // 4,194,304 halves per [B,N,C] tensor
    _Float16* d1T = H;                          // desc1 f16 [b][n][c]
    _Float16* d2T = H + T;                      // desc2 f16
    _Float16* qT  = H + 2 * T;                  // q  -> attn (aliased, per-wave safe)
    _Float16* kT  = H + 3 * T;                  // k  -> m
    _Float16* vT  = H + 4 * T;                  // v  -> y (y spans 4T..6T)
    _Float16* yT  = vT;                         // [b][n][512], 2T halves
    _Float16* wqh = H + 6 * T;
    _Float16* wkh = wqh + CC * CC;
    _Float16* wvh = wkh + CC * CC;
    _Float16* wmh = wvh + CC * CC;
    _Float16* wc1h = wmh + CC * CC;             // 512*512
    _Float16* wc2h = wc1h + 4 * CC * CC;        // 256*512
    float* bnscale = (float*)(wc2h + 2 * CC * CC);
    float* bnshift = bnscale + 2 * CC;

    // ---- pre-convert everything to f16 (once per call) ----
    cvt_transpose_kernel<<<dim3(NN / 32, CC / 32, BB), 256, 0, stream>>>(desc1, d1T, CC, NN);
    cvt_transpose_kernel<<<dim3(NN / 32, CC / 32, BB), 256, 0, stream>>>(desc2, d2T, CC, NN);
    cvt_kernel<<<(CC * CC) / 256, 256, 0, stream>>>(wq, wqh, CC * CC);
    cvt_kernel<<<(CC * CC) / 256, 256, 0, stream>>>(wk, wkh, CC * CC);
    cvt_kernel<<<(CC * CC) / 256, 256, 0, stream>>>(wv, wvh, CC * CC);
    cvt_kernel<<<(CC * CC) / 256, 256, 0, stream>>>(wm, wmh, CC * CC);
    cvt_kernel<<<(4 * CC * CC) / 256, 256, 0, stream>>>(wc1, wc1h, 4 * CC * CC);
    cvt_kernel<<<(2 * CC * CC) / 256, 256, 0, stream>>>(wc2, wc2h, 2 * CC * CC);
    bn_prep_kernel<<<2, 256, 0, stream>>>(bn_gamma, bn_beta, bn_mean, bn_var,
                                          bnscale, bnshift, 2 * CC);

    const dim3 blk(256);
    const dim3 gC(NN / 64, CC / 128, BB);         // 256-row GEMMs (block 128M x 64N)
    const dim3 g2C(NN / 64, (2 * CC) / 128, BB);  // 512-row GEMM

    // q,k,v (f16, [b][n][c])
    gemm_wmma_f16<<<gC, blk, 0, stream>>>(wqh, d1T, d1T, CC, qT, 0, bq, nullptr,
                                          nullptr, CC, CC, NN, 0);
    gemm_wmma_f16<<<gC, blk, 0, stream>>>(wkh, d2T, d2T, CC, kT, 0, bk, nullptr,
                                          nullptr, CC, CC, NN, 0);
    gemm_wmma_f16<<<gC, blk, 0, stream>>>(wvh, d2T, d2T, CC, vT, 0, bv, nullptr,
                                          nullptr, CC, CC, NN, 0);

    // attention: one wave per (b,h,n); output aliases qT region
    attn_kernel<<<(BB * HH * NN) / 8, 256, 0, stream>>>(qT, kT, vT, qT,
                                                        neighb, nmask, res_lam);

    // m = wm * attn + bm   (into kT region, f16 [b][n][c])
    gemm_wmma_f16<<<gC, blk, 0, stream>>>(wmh, qT, qT, CC, kT, 0, bm, nullptr,
                                          nullptr, CC, CC, NN, 0);

    // y = relu(BN(wc1 * [desc1 ; m]))  (f16 [b][n][512], spans vT..vT+2T)
    gemm_wmma_f16<<<g2C, blk, 0, stream>>>(wc1h, d1T, kT, CC, yT, 0, bnshift,
                                           bnscale, nullptr, 2 * CC, 2 * CC, NN, 1);

    // out = desc1 + wc2 * y + bc2   (f32 [b][c][n] to d_out)
    gemm_wmma_f16<<<gC, blk, 0, stream>>>(wc2h, yT, yT, 2 * CC, d_out, 1, bc2,
                                          nullptr, desc1, CC, 2 * CC, NN, 0);
}
